// MultiHeadAttention_28630251995484
// MI455X (gfx1250) — matrix operand order repaired and checked by reading, not compiled, not run on hardware
//
#include <hip/hip_runtime.h>

// ---------------------------------------------------------------------------
// Causal MHA for MI455X / gfx1250.
// bf16 WMMA (v_wmma_f32_16x16x32_bf16), flash-attention streaming softmax,
// M=32 per-wave tiles so every streamed B-fragment feeds 2 WMMAs.
// ---------------------------------------------------------------------------

constexpr int Bn = 8, Tn = 2048, Cn = 512, Hn = 8, Dn = 64;

typedef __attribute__((ext_vector_type(16))) __bf16      v16bf;
typedef __attribute__((ext_vector_type(8)))  float        v8f;
typedef __attribute__((ext_vector_type(4)))  unsigned int u32x4;

union Frag { v16bf bf; u32x4 u[2]; };

__device__ __forceinline__ unsigned short f2bf(float f) {
  unsigned int u = __float_as_uint(f);
  u += 0x7FFFu + ((u >> 16) & 1u);        // round-to-nearest-even
  return (unsigned short)(u >> 16);
}

__device__ __forceinline__ void ldfrag(Frag& f, const unsigned short* p0,
                                       const unsigned short* p1) {
  f.u[0] = *(const u32x4*)p0;             // 16B contiguous
  f.u[1] = *(const u32x4*)p1;             // 16B contiguous
}

__device__ __forceinline__ v8f wmma_bf16(const Frag& a, const Frag& b, v8f c) {
  return __builtin_amdgcn_wmma_f32_16x16x32_bf16(false, a.bf, false, b.bf,
                                                 (short)0, c, false, false);
}

__device__ __forceinline__ v8f zero8() {
  v8f z;
#pragma unroll
  for (int i = 0; i < 8; i++) z[i] = 0.f;
  return z;
}

__device__ __forceinline__ float rmax16(float v) {
  v = fmaxf(v, __shfl_xor(v, 8, 16));
  v = fmaxf(v, __shfl_xor(v, 4, 16));
  v = fmaxf(v, __shfl_xor(v, 2, 16));
  v = fmaxf(v, __shfl_xor(v, 1, 16));
  return v;
}
__device__ __forceinline__ float rsum16(float v) {
  v += __shfl_xor(v, 8, 16);
  v += __shfl_xor(v, 4, 16);
  v += __shfl_xor(v, 2, 16);
  v += __shfl_xor(v, 1, 16);
  return v;
}

// ---------------- conversion / weight transposition ------------------------

__global__ __launch_bounds__(256) void cvt_bf16(const float* __restrict__ in,
                                                unsigned short* __restrict__ out,
                                                int n) {
  int stride = gridDim.x * blockDim.x;
  for (int i = blockIdx.x * blockDim.x + threadIdx.x; i < n; i += stride)
    out[i] = f2bf(in[i]);
}

// in[b][r][c] (f32) -> out[b][c][r] (bf16)   (N-major weights for B-frags)
__global__ __launch_bounds__(256) void cvt_transpose(const float* __restrict__ in,
                                                     unsigned short* __restrict__ out,
                                                     int batch, int rows, int cols) {
  int n = batch * rows * cols;
  int stride = gridDim.x * blockDim.x;
  for (int idx = blockIdx.x * blockDim.x + threadIdx.x; idx < n; idx += stride) {
    int c = idx % cols;
    int t = idx / cols;
    int r = t % rows;
    int bb = t / rows;
    out[((size_t)bb * cols + c) * rows + r] = f2bf(in[idx]);
  }
}

// ---------------- QKV projection: [T,C] @ [C,D] per (mat,b,h) --------------
// One wave computes a 32(M) x 64(N=D) strip; each weight B-frag feeds 2 WMMAs.

__global__ __launch_bounds__(256) void qkv_kernel(
    const unsigned short* __restrict__ xb,
    const unsigned short* __restrict__ WqT, const unsigned short* __restrict__ WkT,
    const unsigned short* __restrict__ WvT,
    unsigned short* __restrict__ qO, unsigned short* __restrict__ kO,
    unsigned short* __restrict__ vTO) {
  const int lane = threadIdx.x & 31;
  const int wave = (int)((blockIdx.x * blockDim.x + threadIdx.x) >> 5);
  const int mt = wave & 63;             // T/32 = 64
  int rest = wave >> 6;
  const int h = rest & 7; rest >>= 3;
  const int b = rest & 7;
  const int mat = rest >> 3;            // 0=q 1=k 2=v
  const unsigned short* WT = (mat == 0) ? WqT : (mat == 1) ? WkT : WvT;
  WT += (size_t)h * Dn * Cn;

  const int half = lane >> 4, l15 = lane & 15;
  const int t0 = mt * 32;
  const unsigned short* abase0 = xb + ((size_t)b * Tn + t0 + l15) * Cn + 8 * half;
  const unsigned short* abase1 = abase0 + (size_t)16 * Cn;

  v8f acc[2][4];
#pragma unroll
  for (int mi = 0; mi < 2; mi++)
#pragma unroll
    for (int j = 0; j < 4; j++) acc[mi][j] = zero8();

  for (int kk = 0; kk < Cn; kk += 32) {
    Frag a0, a1;
    ldfrag(a0, abase0 + kk, abase0 + kk + 16);
    ldfrag(a1, abase1 + kk, abase1 + kk + 16);
#pragma unroll
    for (int j = 0; j < 4; j++) {
      const unsigned short* p = WT + (size_t)(16 * j + l15) * Cn + kk + 8 * half;
      Frag bb;
      ldfrag(bb, p, p + 16);
      acc[0][j] = wmma_bf16(a0, bb, acc[0][j]);
      acc[1][j] = wmma_bf16(a1, bb, acc[1][j]);
    }
  }

  const size_t bh = (size_t)b * Hn + h;
#pragma unroll
  for (int mi = 0; mi < 2; mi++) {
#pragma unroll
    for (int j = 0; j < 4; j++) {
#pragma unroll
      for (int i = 0; i < 8; i++) {
        int t = t0 + 16 * mi + i + 8 * half;
        int d = 16 * j + l15;
        unsigned short val = f2bf(acc[mi][j][i]);
        if (mat == 2)
          vTO[(bh * Dn + d) * Tn + t] = val;        // v stored transposed [d][t]
        else
          ((mat == 0) ? qO : kO)[(bh * Tn + t) * Dn + d] = val;
      }
    }
  }
}

// ---------------- flash attention: one wave = 32 rows x full D -------------

__global__ __launch_bounds__(256) void attn_kernel(
    const unsigned short* __restrict__ qI, const unsigned short* __restrict__ kI,
    const unsigned short* __restrict__ vTI, unsigned short* __restrict__ ao) {
  __shared__ __align__(16) unsigned short ptile[8][32 * 32];

  const int lane = threadIdx.x & 31;
  const int wslot = (int)(threadIdx.x >> 5);
  const int wave = (int)((blockIdx.x * blockDim.x + threadIdx.x) >> 5);
  const int mt = wave & 63;             // T/32 = 64
  int rest = wave >> 6;
  const int h = rest & 7;
  const int b = rest >> 3;
  const size_t bh = (size_t)b * Hn + h;
  const int half = lane >> 4, l15 = lane & 15;
  const int t0 = mt * 32;

  // q A-frags for 32 rows x K=64 (four 16x32 frags), kept resident.
  const unsigned short* qb0 = qI + (bh * Tn + t0 + l15) * Dn + 8 * half;
  const unsigned short* qb1 = qb0 + (size_t)16 * Dn;
  Frag qa[2][2];
  ldfrag(qa[0][0], qb0, qb0 + 16);
  ldfrag(qa[0][1], qb0 + 32, qb0 + 48);
  ldfrag(qa[1][0], qb1, qb1 + 16);
  ldfrag(qa[1][1], qb1 + 32, qb1 + 48);

  v8f oacc[2][4];
#pragma unroll
  for (int mi = 0; mi < 2; mi++)
#pragma unroll
    for (int j = 0; j < 4; j++) oacc[mi][j] = zero8();
  float rmx[2][8], rsm[2][8];
#pragma unroll
  for (int mi = 0; mi < 2; mi++)
#pragma unroll
    for (int i = 0; i < 8; i++) { rmx[mi][i] = -1e30f; rsm[mi][i] = 0.f; }

  const float scale = 0.125f;                 // 1/sqrt(64)
  const int trow = t0 + 8 * half;

  for (int s0 = 0; s0 <= t0 + 31; s0 += 32) {
    // ---- scores: q(32x64) @ k^T(64x32) -> four 16x16 accum tiles ----
    v8f sa[2][2];
#pragma unroll
    for (int mi = 0; mi < 2; mi++) { sa[mi][0] = zero8(); sa[mi][1] = zero8(); }
    const unsigned short* kb0 = kI + (bh * Tn + s0 + l15) * Dn + 8 * half;
    const unsigned short* kb1 = kb0 + 16 * Dn;
    Frag fb;
    ldfrag(fb, kb0, kb0 + 16);
    sa[0][0] = wmma_bf16(qa[0][0], fb, sa[0][0]);
    sa[1][0] = wmma_bf16(qa[1][0], fb, sa[1][0]);
    ldfrag(fb, kb0 + 32, kb0 + 48);
    sa[0][0] = wmma_bf16(qa[0][1], fb, sa[0][0]);
    sa[1][0] = wmma_bf16(qa[1][1], fb, sa[1][0]);
    ldfrag(fb, kb1, kb1 + 16);
    sa[0][1] = wmma_bf16(qa[0][0], fb, sa[0][1]);
    sa[1][1] = wmma_bf16(qa[1][0], fb, sa[1][1]);
    ldfrag(fb, kb1 + 32, kb1 + 48);
    sa[0][1] = wmma_bf16(qa[0][1], fb, sa[0][1]);
    sa[1][1] = wmma_bf16(qa[1][1], fb, sa[1][1]);

    // prefetch next s-block k/v lines (global_prefetch_b8) while softmax runs
    if (s0 + 32 <= t0 + 31) {
      __builtin_prefetch(kb0 + (size_t)32 * Dn, 0, 0);
      __builtin_prefetch(kb1 + (size_t)32 * Dn, 0, 0);
      __builtin_prefetch(vTI + (bh * Dn + l15) * Tn + (s0 + 32) + 16 * half, 0, 0);
    }

    // ---- causal mask + online softmax update ----
    const int c0 = s0 + l15, c1 = c0 + 16;
#pragma unroll
    for (int mi = 0; mi < 2; mi++) {
#pragma unroll
      for (int i = 0; i < 8; i++) {
        const int t = trow + 16 * mi + i;
        float f0 = (c0 <= t) ? sa[mi][0][i] * scale : -1e30f;
        float f1 = (c1 <= t) ? sa[mi][1][i] * scale : -1e30f;
        float bm = rmax16(fmaxf(f0, f1));
        float nm = fmaxf(rmx[mi][i], bm);
        float corr = __expf(rmx[mi][i] - nm);
        rmx[mi][i] = nm;
        float e0 = __expf(f0 - nm);
        float e1 = __expf(f1 - nm);
        rsm[mi][i] = rsm[mi][i] * corr + rsum16(e0 + e1);
#pragma unroll
        for (int j = 0; j < 4; j++) oacc[mi][j][i] *= corr;
        // stage p tile (row-major 32x32 bf16) for A-frag re-read
        ptile[wslot][(16 * mi + i + 8 * half) * 32 + l15]      = f2bf(e0);
        ptile[wslot][(16 * mi + i + 8 * half) * 32 + l15 + 16] = f2bf(e1);
      }
    }
    asm volatile("s_wait_dscnt 0" ::: "memory");   // cross-lane LDS RAW

    Frag pa0, pa1;
    const unsigned short* pb0 = &ptile[wslot][(size_t)l15 * 32 + 8 * half];
    const unsigned short* pb1 = pb0 + 16 * 32;
    ldfrag(pa0, pb0, pb0 + 16);
    ldfrag(pa1, pb1, pb1 + 16);

    // ---- oacc += p(32x32) @ v(32x64) ----
#pragma unroll
    for (int j = 0; j < 4; j++) {
      const unsigned short* vp = vTI + (bh * Dn + 16 * j + l15) * Tn + s0 + 8 * half;
      Frag fv;
      ldfrag(fv, vp, vp + 16);
      oacc[0][j] = wmma_bf16(pa0, fv, oacc[0][j]);
      oacc[1][j] = wmma_bf16(pa1, fv, oacc[1][j]);
    }
    asm volatile("" ::: "memory");                 // keep LDS WAR ordering
  }

  // normalize + write concat-head layout [b][t][h*D + d]
#pragma unroll
  for (int mi = 0; mi < 2; mi++) {
#pragma unroll
    for (int j = 0; j < 4; j++) {
#pragma unroll
      for (int i = 0; i < 8; i++) {
        int t = trow + 16 * mi + i;
        int d = 16 * j + l15;
        ao[((size_t)b * Tn + t) * Cn + h * Dn + d] =
            f2bf(oacc[mi][j][i] / rsm[mi][i]);
      }
    }
  }
}

// ---------------- output projection: [B*T,C] @ [C,C] + bias ----------------

__global__ __launch_bounds__(256) void proj_kernel(
    const unsigned short* __restrict__ ao, const unsigned short* __restrict__ WoT,
    const float* __restrict__ bo, float* __restrict__ out) {
  const int lane = threadIdx.x & 31;
  const int wave = (int)((blockIdx.x * blockDim.x + threadIdx.x) >> 5);
  const int nt = wave & 7;            // C/64 = 8
  const int mt = wave >> 3;           // B*T/32 = 512
  const int half = lane >> 4, l15 = lane & 15;
  const int r0 = mt * 32, n0 = nt * 64;
  const unsigned short* abase0 = ao + (size_t)(r0 + l15) * Cn + 8 * half;
  const unsigned short* abase1 = abase0 + (size_t)16 * Cn;

  v8f acc[2][4];
#pragma unroll
  for (int mi = 0; mi < 2; mi++)
#pragma unroll
    for (int j = 0; j < 4; j++) acc[mi][j] = zero8();

  for (int kk = 0; kk < Cn; kk += 32) {
    Frag a0, a1;
    ldfrag(a0, abase0 + kk, abase0 + kk + 16);
    ldfrag(a1, abase1 + kk, abase1 + kk + 16);
#pragma unroll
    for (int j = 0; j < 4; j++) {
      const unsigned short* p = WoT + (size_t)(n0 + 16 * j + l15) * Cn + kk + 8 * half;
      Frag bb;
      ldfrag(bb, p, p + 16);
      acc[0][j] = wmma_bf16(a0, bb, acc[0][j]);
      acc[1][j] = wmma_bf16(a1, bb, acc[1][j]);
    }
  }

#pragma unroll
  for (int mi = 0; mi < 2; mi++) {
#pragma unroll
    for (int j = 0; j < 4; j++) {
      float bias = bo[n0 + 16 * j + l15];
#pragma unroll
      for (int i = 0; i < 8; i++)
        out[(size_t)(r0 + 16 * mi + i + 8 * half) * Cn + n0 + 16 * j + l15] =
            acc[mi][j][i] + bias;
    }
  }
}

// ---------------------------------------------------------------------------

extern "C" void kernel_launch(void* const* d_in, const int* in_sizes, int n_in,
                              void* d_out, int out_size, void* d_ws, size_t ws_size,
                              hipStream_t stream) {
  (void)in_sizes; (void)n_in; (void)out_size; (void)ws_size;
  const float* x  = (const float*)d_in[0];
  const float* Wq = (const float*)d_in[1];
  const float* Wk = (const float*)d_in[2];
  const float* Wv = (const float*)d_in[3];
  const float* Wo = (const float*)d_in[4];
  const float* bo = (const float*)d_in[5];
  float* out = (float*)d_out;

  // workspace carve-up (bf16 = unsigned short), ~86 MB total
  unsigned short* xb  = (unsigned short*)d_ws;                 // B*T*C
  unsigned short* WqT = xb  + (size_t)Bn * Tn * Cn;            // H*D*C (N-major)
  unsigned short* WkT = WqT + (size_t)Hn * Dn * Cn;
  unsigned short* WvT = WkT + (size_t)Hn * Dn * Cn;
  unsigned short* WoT = WvT + (size_t)Hn * Dn * Cn;            // C*C (N-major)
  unsigned short* qB  = WoT + (size_t)Cn * Cn;                 // [b,h,t,d]
  unsigned short* kB  = qB  + (size_t)Bn * Hn * Tn * Dn;       // [b,h,t,d]
  unsigned short* vTB = kB  + (size_t)Bn * Hn * Tn * Dn;       // [b,h,d,t]
  unsigned short* aoB = vTB + (size_t)Bn * Hn * Tn * Dn;       // [b,t,C]

  cvt_bf16<<<2048, 256, 0, stream>>>(x, xb, Bn * Tn * Cn);
  cvt_transpose<<<512, 256, 0, stream>>>(Wq, WqT, Hn, Cn, Dn);
  cvt_transpose<<<512, 256, 0, stream>>>(Wk, WkT, Hn, Cn, Dn);
  cvt_transpose<<<512, 256, 0, stream>>>(Wv, WvT, Hn, Cn, Dn);
  cvt_transpose<<<512, 256, 0, stream>>>(Wo, WoT, 1, Cn, Cn);

  // 3 * B*H * (T/32) = 12288 waves -> 1536 blocks of 8 waves
  qkv_kernel<<<1536, 256, 0, stream>>>(xb, WqT, WkT, WvT, qB, kB, vTB);
  // B*H*(T/32) = 4096 waves -> 512 blocks
  attn_kernel<<<512, 256, 0, stream>>>(qB, kB, vTB, aoB);
  // (B*T/32)*(C/64) = 4096 waves -> 512 blocks
  proj_kernel<<<512, 256, 0, stream>>>(aoB, WoT, bo, out);
}
